// GRUODEDecay_73993696575832
// MI455X (gfx1250) — matrix operand order repaired and checked by reading, not compiled, not run on hardware
//
#include <hip/hip_runtime.h>
#include <hip/hip_bf16.h>

// ---------------------------------------------------------------------------
// GRU-ODE decay RNN for MI455X (gfx1250), wave32 + WMMA bf16 path.
// B=1024, S=256, I=H=128. One workgroup owns 16 batch rows and runs the whole
// sequential time loop; 8 waves split the 128/384 output columns into 16-wide
// WMMA tiles. GEMMs use v_wmma_f32_16x16x32_bf16 (f32 accumulate).
// x / times for step s+1 are software-pipelined into registers so their
// global-memory latency hides behind step s's compute.
// ---------------------------------------------------------------------------

typedef __bf16 bf16_t;
typedef __attribute__((ext_vector_type(16))) __bf16 v16bf;
typedef __attribute__((ext_vector_type(8)))  __bf16 v8bf;
typedef __attribute__((ext_vector_type(8)))  float  v8f;

constexpr int Bsz = 1024;
constexpr int Ssz = 256;
constexpr int Isz = 128;
constexpr int Hsz = 128;
constexpr int H3  = 384;
constexpr int EULER = 4;

// One 16x16 f32 tile of  A[16x128] @ W^T  for an output column block; K = 128.
// a_row : LDS bf16 row for this lane's A-row (length 128, K-contiguous).
// w_base: bf16 weights, row-major [*, 128]; wrow = global output column.
// Accumulator initialized with the (lane-uniform-per-column) bias.
__device__ __forceinline__ v8f gemm16x128(const bf16_t* __restrict__ a_row,
                                          const bf16_t* __restrict__ w_base,
                                          int wrow, int half, float bias) {
  v8f acc;
#pragma unroll
  for (int i = 0; i < 8; ++i) acc[i] = bias;
  const bf16_t* wr = w_base + (size_t)wrow * Hsz;
#pragma unroll
  for (int kk = 0; kk < 4; ++kk) {
    // A fragment: element j <-> K = kk*32 + (j<8?0:16) + half*8 + (j&7)
    v8bf lo = *(const v8bf*)(a_row + kk * 32 + half * 8);
    v8bf hi = *(const v8bf*)(a_row + kk * 32 + 16 + half * 8);
    v16bf a = __builtin_shufflevector(lo, hi, 0, 1, 2, 3, 4, 5, 6, 7,
                                      8, 9, 10, 11, 12, 13, 14, 15);
    // B fragment: element j <-> K = kk*32 + (j<8?0:16) + half*8 + (j&7)
    v8bf blo = *(const v8bf*)(wr + kk * 32 + half * 8);
    v8bf bhi = *(const v8bf*)(wr + kk * 32 + 16 + half * 8);
    v16bf b = __builtin_shufflevector(blo, bhi, 0, 1, 2, 3, 4, 5, 6, 7,
                                      8, 9, 10, 11, 12, 13, 14, 15);
    acc = __builtin_amdgcn_wmma_f32_16x16x32_bf16(
        /*neg_a=*/false, a, /*neg_b=*/false, b,
        /*c_mod=*/(short)0, acc, /*reuse_a=*/false, /*reuse_b=*/false);
  }
  return acc;
}

__device__ __forceinline__ float sigmoid_f(float x) {
  return 1.0f / (1.0f + __expf(-x));
}

// ---------------------------------------------------------------------------
// Prep: f32 -> bf16 weight conversion
// ---------------------------------------------------------------------------
__global__ void cvt_bf16_kernel(const float* __restrict__ src,
                                bf16_t* __restrict__ dst, int n) {
  int i = blockIdx.x * blockDim.x + threadIdx.x;
  if (i < n) dst[i] = (bf16_t)src[i];
}

// ---------------------------------------------------------------------------
// Prep: per-timestep min over the batch of times[b][s]
// ---------------------------------------------------------------------------
__global__ __launch_bounds__(256) void tmin_kernel(
    const float* __restrict__ times, float* __restrict__ tmin) {
  __shared__ float red[256];
  const int s = blockIdx.x;
  float v = 3.4028235e38f;
  for (int b = threadIdx.x; b < Bsz; b += 256)
    v = fminf(v, times[(size_t)b * Ssz + s]);
  red[threadIdx.x] = v;
  __syncthreads();
  for (int off = 128; off > 0; off >>= 1) {
    if (threadIdx.x < off)
      red[threadIdx.x] = fminf(red[threadIdx.x], red[threadIdx.x + off]);
    __syncthreads();
  }
  if (threadIdx.x == 0) tmin[s] = red[0];
}

// ---------------------------------------------------------------------------
// Main fused sequential kernel: 64 blocks x 256 threads (8 waves)
// ---------------------------------------------------------------------------
__global__ __launch_bounds__(256) void gruode_main_kernel(
    const float* __restrict__ x, const float* __restrict__ times,
    const float* __restrict__ b_ih, const float* __restrict__ b_hh,
    const float* __restrict__ b1, const float* __restrict__ b2,
    const bf16_t* __restrict__ w_ih, const bf16_t* __restrict__ w_hh,
    const bf16_t* __restrict__ w1, const bf16_t* __restrict__ w2,
    const float* __restrict__ tmin, float* __restrict__ out) {
  __shared__ __align__(32) bf16_t xs[16 * 128];  // x tile, bf16 (A staging)
  __shared__ __align__(32) bf16_t hb[16 * 128];  // h state, bf16 (A staging)
  __shared__ __align__(32) bf16_t tb[16 * 128];  // tanh tile, bf16 (A staging)
  __shared__ float hf[16 * 128];                 // h state, f32 (exact carry)
  __shared__ float dts[16];                      // per-row dt for this step

  const int tid  = threadIdx.x;
  const int lane = tid & 31;
  const int wave = tid >> 5;
  const int lcol = lane & 15;   // tile column / A-row index
  const int half = lane >> 4;   // K-half selector for fragments
  const int n0   = wave * 16;   // this wave's output-column block
  const int ncol = n0 + lcol;   // this lane's global output column (0..127)
  const int b0   = blockIdx.x * 16;

  // h0 = 0
  for (int e = tid; e < 16 * 128; e += 256) {
    hf[e] = 0.0f;
    hb[e] = (bf16_t)0.0f;
  }

  // Lane-resident biases (column-indexed; uniform across C-tile rows)
  const float bihr = b_ih[ncol], bihz = b_ih[Hsz + ncol], bihn = b_ih[2 * Hsz + ncol];
  const float bhhr = b_hh[ncol], bhhz = b_hh[Hsz + ncol], bhhn = b_hh[2 * Hsz + ncol];
  const float b1c = b1[ncol], b2c = b2[ncol];

  const bf16_t* arow_x = xs + lcol * 128;
  const bf16_t* arow_h = hb + lcol * 128;
  const bf16_t* arow_t = tb + lcol * 128;

  // ---- Software pipeline: each thread stages 8 contiguous x elements ------
  const int e0  = tid * 8;        // 256 threads x 8 elems = 16*128
  const int row = e0 >> 7;
  const int k   = e0 & 127;
  const float* xrow = x + (size_t)(b0 + row) * (Ssz * Isz) + k;

  float xv[8];
  float tvreg = 0.0f, tmreg = 0.0f;
  {
#pragma unroll
    for (int j = 0; j < 8; ++j) xv[j] = xrow[j];  // x[.., s=0, ..]
    if (tid < 16) {
      tvreg = times[(size_t)(b0 + tid) * Ssz];
      tmreg = tmin[0];
    }
  }
  __syncthreads();  // covers h0 init too

#pragma unroll 1
  for (int s = 0; s < Ssz; ++s) {
    // ---- Commit pipelined registers into LDS (bf16) + per-row dt ----------
#pragma unroll
    for (int j = 0; j < 8; ++j) xs[row * 128 + k + j] = (bf16_t)xv[j];
    if (tid < 16) dts[tid] = (tvreg - tmreg) * (1.0f / EULER);
    __syncthreads();

    // ---- Issue next step's loads now; consumed next iteration -------------
    if (s + 1 < Ssz) {
      const float* p = xrow + (size_t)(s + 1) * Isz;
#pragma unroll
      for (int j = 0; j < 8; ++j) xv[j] = p[j];
      if (tid < 16) {
        tvreg = times[(size_t)(b0 + tid) * Ssz + (s + 1)];
        tmreg = tmin[s + 1];
      }
    }

    // ---- GRU gates: 6 WMMA tile-chains fully in registers ------------------
    v8f gir = gemm16x128(arow_x, w_ih, ncol,           half, bihr);
    v8f giz = gemm16x128(arow_x, w_ih, Hsz + ncol,     half, bihz);
    v8f gin = gemm16x128(arow_x, w_ih, 2 * Hsz + ncol, half, bihn);
    v8f ghr = gemm16x128(arow_h, w_hh, ncol,           half, bhhr);
    v8f ghz = gemm16x128(arow_h, w_hh, Hsz + ncol,     half, bhhz);
    v8f ghn = gemm16x128(arow_h, w_hh, 2 * Hsz + ncol, half, bhhn);
    __syncthreads();  // all waves finished reading hb/xs before hb overwrite

#pragma unroll
    for (int r = 0; r < 8; ++r) {
      const int m = r + 8 * half;  // C/D row for this VGPR slot
      float rr = sigmoid_f(gir[r] + ghr[r]);
      float zz = sigmoid_f(giz[r] + ghz[r]);
      float nn = tanhf(gin[r] + rr * ghn[r]);
      float hp = hf[m * 128 + ncol];
      float hn = (1.0f - zz) * nn + zz * hp;
      out[(size_t)(b0 + m) * (Ssz * Hsz) + (size_t)s * Hsz + ncol] = hn;
      hf[m * 128 + ncol] = hn;           // pre-decay state -> Euler start
      hb[m * 128 + ncol] = (bf16_t)hn;
    }
    __syncthreads();

    // ---- ODE decay: 4 fixed-step Euler iterations of f(h)=W2 tanh(W1 h) ----
#pragma unroll 1
    for (int e = 0; e < EULER; ++e) {
      v8f t1 = gemm16x128(arow_h, w1, ncol, half, b1c);
#pragma unroll
      for (int r = 0; r < 8; ++r) {
        const int m = r + 8 * half;
        tb[m * 128 + ncol] = (bf16_t)tanhf(t1[r]);
      }
      __syncthreads();  // tb ready for all waves

      v8f f = gemm16x128(arow_t, w2, ncol, half, b2c);
#pragma unroll
      for (int r = 0; r < 8; ++r) {
        const int m = r + 8 * half;
        float hd = hf[m * 128 + ncol] + dts[m] * f[r];
        hf[m * 128 + ncol] = hd;
        hb[m * 128 + ncol] = (bf16_t)hd;
      }
      __syncthreads();  // hb/hf stable before next gemm1 / next step
    }
  }
}

// ---------------------------------------------------------------------------
// Launch
// ---------------------------------------------------------------------------
extern "C" void kernel_launch(void* const* d_in, const int* in_sizes, int n_in,
                              void* d_out, int out_size, void* d_ws,
                              size_t ws_size, hipStream_t stream) {
  (void)in_sizes; (void)n_in; (void)out_size; (void)ws_size;
  const float* x     = (const float*)d_in[0];
  const float* times = (const float*)d_in[1];
  const float* w_ih  = (const float*)d_in[2];
  const float* w_hh  = (const float*)d_in[3];
  const float* b_ih  = (const float*)d_in[4];
  const float* b_hh  = (const float*)d_in[5];
  const float* w1    = (const float*)d_in[6];
  const float* b1    = (const float*)d_in[7];
  const float* w2    = (const float*)d_in[8];
  const float* b2    = (const float*)d_in[9];
  float* out = (float*)d_out;

  // Workspace layout (bytes): bf16 weights + tmin. Total ~257 KB.
  char* ws = (char*)d_ws;
  bf16_t* wihb = (bf16_t*)(ws + 0);        // 384*128*2 = 98304
  bf16_t* whhb = (bf16_t*)(ws + 98304);    // 98304
  bf16_t* w1b  = (bf16_t*)(ws + 196608);   // 128*128*2 = 32768
  bf16_t* w2b  = (bf16_t*)(ws + 229376);   // 32768
  float*  tmin = (float*)(ws + 262144);    // 256*4

  cvt_bf16_kernel<<<(H3 * Hsz + 255) / 256, 256, 0, stream>>>(w_ih, wihb, H3 * Hsz);
  cvt_bf16_kernel<<<(H3 * Hsz + 255) / 256, 256, 0, stream>>>(w_hh, whhb, H3 * Hsz);
  cvt_bf16_kernel<<<(Hsz * Hsz + 255) / 256, 256, 0, stream>>>(w1, w1b, Hsz * Hsz);
  cvt_bf16_kernel<<<(Hsz * Hsz + 255) / 256, 256, 0, stream>>>(w2, w2b, Hsz * Hsz);
  tmin_kernel<<<Ssz, 256, 0, stream>>>(times, tmin);

  gruode_main_kernel<<<Bsz / 16, 256, 0, stream>>>(
      x, times, b_ih, b_hh, b1, b2, wihb, whhb, w1b, w2b, tmin, out);
}
